// Decoder_57930518888989
// MI455X (gfx1250) — matrix operand order repaired and checked by reading, not compiled, not run on hardware
//
#include <hip/hip_runtime.h>
#include <hip/hip_bf16.h>
#include <math.h>

// ---------------- problem constants ----------------
#define BB 64
#define SS 128
#define TT 128
#define EE 1024
#define HH 1024
#define NEGINF (-1.0e9f)

typedef __bf16 bf16_t;
typedef __attribute__((ext_vector_type(16))) __bf16 bf16x16;
typedef __attribute__((ext_vector_type(8)))  __bf16 bf16x8;
typedef __attribute__((ext_vector_type(8)))  float  f32x8;

// ---------------- fp32 -> bf16 conversion ----------------
__global__ __launch_bounds__(256) void cvt_f32_bf16(const float* __restrict__ in,
                                                    bf16_t* __restrict__ out, int n) {
    int i = blockIdx.x * 256 + threadIdx.x;
    if (i < n) out[i] = (bf16_t)in[i];
}

// ---------------- generic WMMA GEMM:  C[m,n] = act( sum_k A[m,k]*W[n,k] + init ) ----
// A: bf16 (M x K, row stride lda).  W: bf16 (N x K, row stride ldw) -> implicit W.T.
// init = bias[n] (optional) + Cin[m*ldcin + n] (optional).
// act: 0 = none, 1 = tanh.  Optional bf16 mirror of C (stride ldcbf).
// One wave computes a 64x16 tile: 4 A-fragments share one B-fragment (4x W reuse).
// K-loop: unroll-by-2 ping-pong double buffering -> loads overlap WMMAs with
// no register-rotation copies.  Requires K % 64 == 0 (true here: 1024/2048/3072).
// Grid: (M/64)*(N/16) waves, 8 waves (256 thr) per block.
__global__ __launch_bounds__(256) void gemm_bf16_wmma(
    const bf16_t* __restrict__ A, int lda,
    const bf16_t* __restrict__ W, int ldw,
    float* __restrict__ C, int ldc,
    int M, int N, int K,
    const float* __restrict__ bias,
    const float* __restrict__ Cin, int ldcin,
    int act,
    bf16_t* __restrict__ Cbf, int ldcbf)
{
    const int tid  = threadIdx.x;
    const int wave = tid >> 5;
    const int lane = tid & 31;
    const int gw   = blockIdx.x * 8 + wave;
    const int nTiles = N >> 4;
    const int n0 = (gw % nTiles) << 4;
    const int m0 = (gw / nTiles) << 6;
    if (m0 >= M) return;                 // wave-uniform

    const int half = lane >> 4;          // K-half select (ISA lane split)
    const int lr   = lane & 15;
    const int n    = n0 + lr;

    // accumulator init (linear, so pre-add bias / hoisted partial sums)
    f32x8 acc[4];
    const float binit = bias ? bias[n] : 0.0f;
#pragma unroll
    for (int mi = 0; mi < 4; ++mi) {
#pragma unroll
        for (int r = 0; r < 8; ++r) {
            const int m = m0 + mi * 16 + half * 8 + r;
            float v = binit;
            if (Cin) v += Cin[(size_t)m * ldcin + n];
            acc[mi][r] = v;
        }
    }

    const bf16_t* __restrict__ wrow = W + (size_t)n * ldw + half * 16;
    const bf16_t* __restrict__ arow = A + (size_t)(m0 + lr) * lda + half * 16;

    // fragment loader for K-block kb
    auto load_frags = [&](int kb, bf16x16& bfr, bf16x16* afr) {
        // B fragment (32x16): lane<16 -> W[n, kb..kb+15]; lane>=16 -> W[n, kb+16..kb+31]
        const bf16_t* wp = wrow + kb;
        bf16x8 blo = *(const bf16x8*)(wp);
        bf16x8 bhi = *(const bf16x8*)(wp + 8);
        bfr = __builtin_shufflevector(blo, bhi,
            0,1,2,3,4,5,6,7,8,9,10,11,12,13,14,15);
#pragma unroll
        for (int mi = 0; mi < 4; ++mi) {
            // A fragment (16x32): lane<16 -> K [kb..kb+15]; lane>=16 -> K [kb+16..kb+31]
            const bf16_t* ap = arow + (size_t)(mi * 16) * lda + kb;
            bf16x8 alo = *(const bf16x8*)(ap);
            bf16x8 ahi = *(const bf16x8*)(ap + 8);
            afr[mi] = __builtin_shufflevector(alo, ahi,
                0,1,2,3,4,5,6,7,8,9,10,11,12,13,14,15);
        }
    };
    auto compute = [&](const bf16x16& bfr, const bf16x16* afr) {
#pragma unroll
        for (int mi = 0; mi < 4; ++mi)
            acc[mi] = __builtin_amdgcn_wmma_f32_16x16x32_bf16(
                false, afr[mi], false, bfr, (short)0, acc[mi], false, false);
    };

    // ping-pong double buffer, unrolled by 2 (K/32 is even)
    bf16x16 b0, b1, a0[4], a1[4];
    load_frags(0, b0, a0);
    load_frags(32, b1, a1);
    const int steady = K / 64 - 1;       // steady-state double-iterations
    int kb = 64;
    for (int it = 0; it < steady; ++it, kb += 64) {
        compute(b0, a0);
        load_frags(kb, b0, a0);          // refill buffer 0 while buffer 1 computes
        compute(b1, a1);
        load_frags(kb + 32, b1, a1);     // refill buffer 1 while buffer 0 computes
    }
    compute(b0, a0);
    compute(b1, a1);

    // epilogue: C/D layout -> (m = half*8 + r, n = lane&15)
#pragma unroll
    for (int mi = 0; mi < 4; ++mi) {
#pragma unroll
        for (int r = 0; r < 8; ++r) {
            const int m = m0 + mi * 16 + half * 8 + r;
            float v = acc[mi][r];
            if (act == 1) v = tanhf(v);
            C[(size_t)m * ldc + n] = v;
            if (Cbf) Cbf[(size_t)m * ldcbf + n] = (bf16_t)v;
        }
    }
}

// ---------------- attention: scores -> softmax -> context (one block per batch b) ----
__global__ __launch_bounds__(256) void attn_kernel(
    const float* __restrict__ q,          // (B,H)
    const float* __restrict__ pk,         // (B,S,H) proj_key
    const unsigned char* __restrict__ mask, // (B,S) bool
    const float* __restrict__ enc,        // (B,S,2H) fp32
    const float* __restrict__ We,         // (H,)
    bf16_t* __restrict__ xcat)            // (B, H+2H): context -> cols [H, 3H)
{
    const int b = blockIdx.x;
    const int tid = threadIdx.x;
    const int wave = tid >> 5, lane = tid & 31;

    __shared__ float sq[HH];
    __shared__ float swe[HH];
    __shared__ float salpha[SS];

    for (int h = tid; h < HH; h += 256) { sq[h] = q[b * HH + h]; swe[h] = We[h]; }
    __syncthreads();

    // energies: 8 waves x 16 s-values
    for (int si = 0; si < 16; ++si) {
        const int s = wave * 16 + si;
        const float* pks = pk + ((size_t)b * SS + s) * HH;
        float part = 0.0f;
        for (int h = lane; h < HH; h += 32)
            part += tanhf(sq[h] + pks[h]) * swe[h];
        for (int off = 16; off; off >>= 1) part += __shfl_xor(part, off, 32);
        if (lane == 0)
            salpha[s] = mask[b * SS + s] ? part : NEGINF;
    }
    __syncthreads();

    // softmax over S=128 (wave 0)
    if (tid < 32) {
        float mx = NEGINF;
        for (int s = tid; s < SS; s += 32) mx = fmaxf(mx, salpha[s]);
        for (int off = 16; off; off >>= 1) mx = fmaxf(mx, __shfl_xor(mx, off, 32));
        float sum = 0.0f;
        for (int s = tid; s < SS; s += 32) { float e = expf(salpha[s] - mx); salpha[s] = e; sum += e; }
        for (int off = 16; off; off >>= 1) sum += __shfl_xor(sum, off, 32);
        const float inv = 1.0f / sum;
        for (int s = tid; s < SS; s += 32) salpha[s] *= inv;
    }
    __syncthreads();

    // context[b,d] = sum_s alpha[s] * enc[b,s,d]   (d in [0,2H))
    for (int d = tid; d < 2 * HH; d += 256) {
        const float* e = enc + (size_t)b * SS * 2 * HH + d;
        float acc = 0.0f;
        for (int s = 0; s < SS; ++s) acc += salpha[s] * e[(size_t)s * 2 * HH];
        xcat[(size_t)b * (3 * HH) + HH + d] = (bf16_t)acc;
    }
}

// ---------------- GRU gates + state update ----------------
__global__ __launch_bounds__(256) void gates_kernel(
    const float* __restrict__ GI,   // (B,3H) includes b_ih + x/ctx parts
    const float* __restrict__ GH,   // (B,3H) includes b_hh
    float* __restrict__ h,          // (B,H) fp32 state (updated)
    bf16_t* __restrict__ xcat,      // (B,3H): h -> cols [0,H)
    float* __restrict__ out_states, // decoder_states (B,T,H)
    int t)
{
    const int idx = blockIdx.x * 256 + threadIdx.x;  // B*H threads
    const int b = idx >> 10, j = idx & 1023;
    const float* gi = GI + (size_t)b * 3 * HH;
    const float* gh = GH + (size_t)b * 3 * HH;
    const float r = 1.0f / (1.0f + expf(-(gi[j] + gh[j])));
    const float z = 1.0f / (1.0f + expf(-(gi[HH + j] + gh[HH + j])));
    const float nn = tanhf(gi[2 * HH + j] + r * gh[2 * HH + j]);
    const float hn = (1.0f - z) * nn + z * h[(size_t)b * HH + j];
    h[(size_t)b * HH + j] = hn;
    xcat[(size_t)b * 3 * HH + j] = (bf16_t)hn;
    out_states[((size_t)b * TT + t) * HH + j] = hn;
}

// ---------------- host side ----------------
static inline size_t ws_take(size_t& off, size_t bytes) {
    off = (off + 255) & ~(size_t)255;
    size_t p = off; off += bytes; return p;
}

static inline void launch_gemm(hipStream_t s,
    const bf16_t* A, int lda, const bf16_t* W, int ldw,
    float* C, int ldc, int M, int N, int K,
    const float* bias, const float* Cin, int ldcin,
    int act, bf16_t* Cbf, int ldcbf)
{
    int waves = (M / 64) * (N / 16);
    hipLaunchKernelGGL(gemm_bf16_wmma, dim3(waves / 8), dim3(256), 0, s,
        A, lda, W, ldw, C, ldc, M, N, K, bias, Cin, ldcin, act, Cbf, ldcbf);
}

extern "C" void kernel_launch(void* const* d_in, const int* in_sizes, int n_in,
                              void* d_out, int out_size, void* d_ws, size_t ws_size,
                              hipStream_t stream) {
    const float* trg_embed  = (const float*)d_in[0];   // (B,T,E)
    const float* enc_hidden = (const float*)d_in[1];   // (B,S,2H)
    const float* enc_final  = (const float*)d_in[2];   // (1,B,2H)
    const unsigned char* src_mask = (const unsigned char*)d_in[3]; // (B,1,S) bool
    /* d_in[4] trg_mask unused */
    const float* W_bridge = (const float*)d_in[5];     // (H,2H)
    const float* b_bridge = (const float*)d_in[6];
    const float* W_key    = (const float*)d_in[7];     // (H,2H)
    const float* W_query  = (const float*)d_in[8];     // (H,H)
    const float* W_energy = (const float*)d_in[9];     // (1,H)
    const float* W_ih     = (const float*)d_in[10];    // (3H, E+2H)
    const float* W_hh     = (const float*)d_in[11];    // (3H, H)
    const float* b_ih     = (const float*)d_in[12];
    const float* b_hh     = (const float*)d_in[13];
    const float* W_pre    = (const float*)d_in[14];    // (2H, E+3H)

    float* out = (float*)d_out;
    float* out_states = out;                                   // (B,T,H)
    float* out_hidden = out + (size_t)BB * TT * HH;            // (1,B,H)
    float* out_pre    = out_hidden + (size_t)BB * HH;          // (B,T,2H)

    // ---- workspace layout ----
    char* ws = (char*)d_ws; size_t off = 0;
    bf16_t* wq_bf    = (bf16_t*)(ws + ws_take(off, (size_t)HH * HH * 2));
    bf16_t* wkey_bf  = (bf16_t*)(ws + ws_take(off, (size_t)HH * 2 * HH * 2));
    bf16_t* wbr_bf   = (bf16_t*)(ws + ws_take(off, (size_t)HH * 2 * HH * 2));
    bf16_t* wih_bf   = (bf16_t*)(ws + ws_take(off, (size_t)3 * HH * 3 * HH * 2));
    bf16_t* whh_bf   = (bf16_t*)(ws + ws_take(off, (size_t)3 * HH * HH * 2));
    bf16_t* wpre_bf  = (bf16_t*)(ws + ws_take(off, (size_t)2 * HH * 4 * HH * 2));
    bf16_t* encf_bf  = (bf16_t*)(ws + ws_take(off, (size_t)BB * 2 * HH * 2));
    bf16_t* ench_bf  = (bf16_t*)(ws + ws_take(off, (size_t)BB * SS * 2 * HH * 2));
    bf16_t* trg_bf   = (bf16_t*)(ws + ws_take(off, (size_t)BB * TT * EE * 2));
    bf16_t* xcat_bf  = (bf16_t*)(ws + ws_take(off, (size_t)BB * 3 * HH * 2)); // [h | context]
    float*  proj_key = (float*) (ws + ws_take(off, (size_t)BB * SS * HH * 4));
    float*  GIX      = (float*) (ws + ws_take(off, (size_t)BB * TT * 3 * HH * 4));
    float*  PREX     = (float*) (ws + ws_take(off, (size_t)BB * TT * 2 * HH * 4));
    float*  h_st     = (float*) (ws + ws_take(off, (size_t)BB * HH * 4));
    float*  q_buf    = (float*) (ws + ws_take(off, (size_t)BB * HH * 4));
    float*  GI       = (float*) (ws + ws_take(off, (size_t)BB * 3 * HH * 4));
    float*  GH       = (float*) (ws + ws_take(off, (size_t)BB * 3 * HH * 4));
    (void)ws_size; (void)in_sizes; (void)n_in; (void)out_size;

    // ---- fp32 -> bf16 conversions (once per call, deterministic) ----
    struct { const float* src; bf16_t* dst; int n; } cv[] = {
        { W_query,   wq_bf,   HH * HH },
        { W_key,     wkey_bf, HH * 2 * HH },
        { W_bridge,  wbr_bf,  HH * 2 * HH },
        { W_ih,      wih_bf,  3 * HH * 3 * HH },
        { W_hh,      whh_bf,  3 * HH * HH },
        { W_pre,     wpre_bf, 2 * HH * 4 * HH },
        { enc_final, encf_bf, BB * 2 * HH },
        { enc_hidden,ench_bf, BB * SS * 2 * HH },
        { trg_embed, trg_bf,  BB * TT * EE },
    };
    for (auto& c : cv)
        hipLaunchKernelGGL(cvt_f32_bf16, dim3((c.n + 255) / 256), dim3(256), 0, stream,
                           c.src, c.dst, c.n);

    // ---- hoisted parallel GEMMs ----
    // h0 = tanh(enc_final @ W_bridge.T + b_bridge); bf16 mirror into xcat[:, 0:H]
    launch_gemm(stream, encf_bf, 2 * HH, wbr_bf, 2 * HH, h_st, HH,
                BB, HH, 2 * HH, b_bridge, nullptr, 0, 1, xcat_bf, 3 * HH);
    // proj_key = enc_hidden @ W_key.T   (M = B*S)
    launch_gemm(stream, ench_bf, 2 * HH, wkey_bf, 2 * HH, proj_key, HH,
                BB * SS, HH, 2 * HH, nullptr, nullptr, 0, 0, nullptr, 0);
    // GIX = trg_embed @ W_ih[:, :E].T + b_ih   (M = B*T)
    launch_gemm(stream, trg_bf, EE, wih_bf, 3 * HH, GIX, 3 * HH,
                BB * TT, 3 * HH, EE, b_ih, nullptr, 0, 0, nullptr, 0);
    // PREX = trg_embed @ W_pre[:, :E].T
    launch_gemm(stream, trg_bf, EE, wpre_bf, 4 * HH, PREX, 2 * HH,
                BB * TT, 2 * HH, EE, nullptr, nullptr, 0, 0, nullptr, 0);

    // ---- sequential decode loop ----
    for (int t = 0; t < TT; ++t) {
        // q = h @ W_query.T
        launch_gemm(stream, xcat_bf, 3 * HH, wq_bf, HH, q_buf, HH,
                    BB, HH, HH, nullptr, nullptr, 0, 0, nullptr, 0);
        // attention -> context (bf16 into xcat[:, H:3H])
        hipLaunchKernelGGL(attn_kernel, dim3(BB), dim3(256), 0, stream,
                           q_buf, proj_key, src_mask, enc_hidden, W_energy, xcat_bf);
        // GI = GIX[:,t,:] + context @ W_ih[:, E:].T
        launch_gemm(stream, xcat_bf + HH, 3 * HH, wih_bf + EE, 3 * HH, GI, 3 * HH,
                    BB, 3 * HH, 2 * HH, nullptr, GIX + (size_t)t * 3 * HH, TT * 3 * HH,
                    0, nullptr, 0);
        // GH = h @ W_hh.T + b_hh
        launch_gemm(stream, xcat_bf, 3 * HH, whh_bf, HH, GH, 3 * HH,
                    BB, 3 * HH, HH, b_hh, nullptr, 0, 0, nullptr, 0);
        // GRU gates -> h_new (fp32 state, bf16 mirror, decoder_states)
        hipLaunchKernelGGL(gates_kernel, dim3(BB * HH / 256), dim3(256), 0, stream,
                           GI, GH, h_st, xcat_bf, out_states, t);
        // pre = PREX[:,t,:] + [h_new | context] @ W_pre[:, E:].T
        launch_gemm(stream, xcat_bf, 3 * HH, wpre_bf + EE, 4 * HH,
                    out_pre + (size_t)t * 2 * HH, TT * 2 * HH,
                    BB, 2 * HH, 3 * HH, nullptr, PREX + (size_t)t * 2 * HH, TT * 2 * HH,
                    0, nullptr, 0);
    }

    // hidden_final = h (last state)
    hipMemcpyAsync(out_hidden, h_st, (size_t)BB * HH * sizeof(float),
                   hipMemcpyDeviceToDevice, stream);
}
